// TreeRNN_33217277067357
// MI455X (gfx1250) — matrix operand order repaired and checked by reading, not compiled, not run on hardware
//
#include <hip/hip_runtime.h>
#include <hip/hip_bf16.h>

typedef __attribute__((ext_vector_type(16))) __bf16 v16bf;
typedef __attribute__((ext_vector_type(8)))  float  v8f;

#define SEQLEN   768
#define BATCH    64
#define NHID     1024
#define K2       2048            // concat width
#define DEPTH    16              // balanced 256-leaf tree needs <= 9
#define BPW      16              // batch elements per workgroup (fills WMMA M dim)
#define NWG      (BATCH / BPW)   // 4 workgroups
#define KBLK     (K2 / 32)       // 64 K-steps of 32
#define NTILE    (NHID / 16)     // 64 N-tiles of 16
#define OPEN_TOK  0
#define CLOSE_TOK 1
#define PAD_TOK   31999          // NTOKEN-1

#define BPACK_ELEMS (KBLK * NTILE * 32 * 16)  // 2,097,152 bf16 = 4 MB

// ---------------------------------------------------------------------------
// Prep: repack W[1024][2048] f32 into fragment-ordered bf16 B-matrix tiles.
// Fragment (kb, nt): lane l, element e -> B[k][n] with
//   k = kb*32 + (l>=16 ? 16 : 0) + e,  n = nt*16 + (l&15)
// and B[k][n] = W[n][k]  (since out = concat @ W.T).
// Each lane's 16 bf16 values are 32 contiguous bytes -> coalesced loads.
// ---------------------------------------------------------------------------
__global__ void pack_w_kernel(const float* __restrict__ W,
                              __bf16* __restrict__ Bp) {
    int id = blockIdx.x * blockDim.x + threadIdx.x;
    if (id >= BPACK_ELEMS) return;
    int e    = id & 15;
    int lane = (id >> 4) & 31;
    int nt   = (id >> 9) & 63;
    int kb   = id >> 15;
    int k = kb * 32 + ((lane >> 4) << 4) + e;
    int n = nt * 16 + (lane & 15);
    Bp[id] = (__bf16)W[(size_t)n * K2 + k];
}

// ---------------------------------------------------------------------------
// Main: one workgroup of 16 wave32s handles 16 batch elements' stack machines.
// Reduces become [16,2048]x[2048,1024] bf16 WMMA GEMMs (f32 accumulate).
// ---------------------------------------------------------------------------
__global__ void tree_rnn_kernel(const int*   __restrict__ tokens,  // [S][B]
                                const float* __restrict__ emb,     // [NTOKEN][NHID]
                                const float* __restrict__ bias,    // [NHID]
                                const __bf16* __restrict__ Bp,     // packed W
                                float* __restrict__ stacks,        // [B][DEPTH][NHID]
                                float* __restrict__ out) {         // [B][NHID]
    __shared__ int s_act[BPW];   // 0 none, 1 shift, 2 close
    __shared__ int s_dst[BPW];   // target stack slot
    __shared__ int s_tok[BPW];
    __shared__ int s_ptr[BPW];
    extern __shared__ __bf16 Abuf[];  // [BPW][K2] bf16 = 64 KB staging

    const int tid  = threadIdx.x;
    const int lane = tid & 31;
    const int wave = tid >> 5;        // 0..15
    const int b0   = blockIdx.x * BPW;

    if (tid < BPW) s_ptr[tid] = 0;
    __syncthreads();

    const int mA    = lane & 15;               // A-matrix row for this lane
    const int kOffA = (lane >> 4) ? 16 : 0;    // contiguous k = 16h + s, matches packed B
    const int nt0   = wave * 4;                // 4 N-tiles per wave
    const int nBase = lane & 15;               // D-tile column for this lane

    for (int t = 0; t < SEQLEN; ++t) {
        // ---- per-batch-element control (uniform via LDS) ----
        if (tid < BPW) {
            int tok = tokens[t * BATCH + b0 + tid];
            int p = s_ptr[tid];
            int act = 0, dst = 0;
            if (tok == CLOSE_TOK)                          { act = 2; dst = p - 2; s_ptr[tid] = p - 1; }
            else if (tok != OPEN_TOK && tok != PAD_TOK)    { act = 1; dst = p;     s_ptr[tid] = p + 1; }
            s_act[tid] = act; s_dst[tid] = dst; s_tok[tid] = tok;
        }
        __syncthreads();

        bool anyShift = false, anyClose = false;
        for (int m = 0; m < BPW; ++m) {
            anyShift |= (s_act[m] == 1);
            anyClose |= (s_act[m] == 2);
        }

        // ---- shifts: push embedding rows (f32) ----
        if (anyShift) {
            for (int m = 0; m < BPW; ++m) {
                if (s_act[m] != 1) continue;
                const float* er = emb + (size_t)s_tok[m] * NHID;
                float* dp = stacks + ((size_t)(b0 + m) * DEPTH + s_dst[m]) * NHID;
                for (int i = tid; i < NHID; i += blockDim.x) dp[i] = er[i];
            }
        }

        // ---- reduces: batched WMMA GEMM ----
        if (anyClose) {
            // Stage concat rows (left at slot dst, right at dst+1: contiguous
            // 2048 f32) into LDS as bf16; zero non-closing rows.
            for (int m = 0; m < BPW; ++m) {
                __bf16* arow = Abuf + m * K2;
                if (s_act[m] == 2) {
                    const float* lp = stacks +
                        ((size_t)(b0 + m) * DEPTH + s_dst[m]) * NHID;
                    for (int i = tid; i < K2; i += blockDim.x)
                        arow[i] = (__bf16)lp[i];
                } else {
                    for (int i = tid; i < K2; i += blockDim.x)
                        arow[i] = (__bf16)0.0f;
                }
            }
            __syncthreads();

            v8f acc0 = {}; v8f acc1 = {}; v8f acc2 = {}; v8f acc3 = {};
            for (int kb = 0; kb < KBLK; ++kb) {
                // A fragment from LDS (two ds_load_b128, contiguous k = 16h + s, as packed B)
                union { v16bf v; uint4 q[2]; } ua;
                const __bf16* ap = Abuf + mA * K2 + kb * 32 + kOffA;
                ua.q[0] = *(const uint4*)(ap);
                ua.q[1] = *(const uint4*)(ap + 8);

                // 4 B fragments: 32 contiguous bytes per lane, stride 512 elems
                const __bf16* bbase =
                    Bp + (((size_t)kb * NTILE + nt0) * 32 + lane) * 16;
                if (kb + 1 < KBLK)
                    __builtin_prefetch(
                        Bp + (((size_t)(kb + 1) * NTILE + nt0) * 32 + lane) * 16,
                        0, 0);
                v16bf bv0 = *(const v16bf*)(bbase);
                v16bf bv1 = *(const v16bf*)(bbase + 512);
                v16bf bv2 = *(const v16bf*)(bbase + 1024);
                v16bf bv3 = *(const v16bf*)(bbase + 1536);

                acc0 = __builtin_amdgcn_wmma_f32_16x16x32_bf16(
                    false, ua.v, false, bv0, (short)0, acc0, false, false);
                acc1 = __builtin_amdgcn_wmma_f32_16x16x32_bf16(
                    false, ua.v, false, bv1, (short)0, acc1, false, false);
                acc2 = __builtin_amdgcn_wmma_f32_16x16x32_bf16(
                    false, ua.v, false, bv2, (short)0, acc2, false, false);
                acc3 = __builtin_amdgcn_wmma_f32_16x16x32_bf16(
                    false, ua.v, false, bv3, (short)0, acc3, false, false);
            }

            // Epilogue: bias + tanh, scatter D tiles (lane L, vgpr r ->
            // row M = r + (L>=16 ? 8 : 0), col N = L&15) into left slot.
            const float bb0 = bias[(nt0 + 0) * 16 + nBase];
            const float bb1 = bias[(nt0 + 1) * 16 + nBase];
            const float bb2 = bias[(nt0 + 2) * 16 + nBase];
            const float bb3 = bias[(nt0 + 3) * 16 + nBase];
            const int mHi = (lane >> 4) ? 8 : 0;
            #pragma unroll
            for (int r = 0; r < 8; ++r) {
                int m = mHi + r;
                if (s_act[m] == 2) {
                    float* dp = stacks +
                        ((size_t)(b0 + m) * DEPTH + s_dst[m]) * NHID;
                    dp[(nt0 + 0) * 16 + nBase] = tanhf(acc0[r] + bb0);
                    dp[(nt0 + 1) * 16 + nBase] = tanhf(acc1[r] + bb1);
                    dp[(nt0 + 2) * 16 + nBase] = tanhf(acc2[r] + bb2);
                    dp[(nt0 + 3) * 16 + nBase] = tanhf(acc3[r] + bb3);
                }
            }
        }

        __threadfence();   // stack writes visible to all waves in the WG
        __syncthreads();
    }

    // Root representation = stack slot 0.
    for (int m = 0; m < BPW; ++m) {
        const float* sp = stacks + (size_t)(b0 + m) * DEPTH * NHID;
        float* op = out + (size_t)(b0 + m) * NHID;
        for (int i = tid; i < NHID; i += blockDim.x) op[i] = sp[i];
    }
}

// ---------------------------------------------------------------------------
// Launch: inputs are (tokens, emb, W, b); out = [64][1024] f32.
// ws layout: [0, 4 MB) packed bf16 W, [4 MB, 8 MB) f32 stacks.
// ---------------------------------------------------------------------------
extern "C" void kernel_launch(void* const* d_in, const int* in_sizes, int n_in,
                              void* d_out, int out_size, void* d_ws, size_t ws_size,
                              hipStream_t stream) {
    const int*   tokens = (const int*)d_in[0];
    const float* emb    = (const float*)d_in[1];
    const float* W      = (const float*)d_in[2];
    const float* b      = (const float*)d_in[3];
    float*       out    = (float*)d_out;

    __bf16* Bp     = (__bf16*)d_ws;
    float*  stacks = (float*)((char*)d_ws + (size_t)BPACK_ELEMS * sizeof(__bf16));

    pack_w_kernel<<<BPACK_ELEMS / 512, 512, 0, stream>>>(W, Bp);

    const size_t ldsBytes = (size_t)BPW * K2 * sizeof(__bf16);  // 64 KB dynamic
    tree_rnn_kernel<<<NWG, 512, ldsBytes, stream>>>(tokens, emb, b, Bp,
                                                    stacks, out);
}
